// FirstLayerAggregator_55362128445578
// MI455X (gfx1250) — matrix operand order repaired and checked by reading, not compiled, not run on hardware
//
#include <hip/hip_runtime.h>
#include <hip/hip_bf16.h>
#include <cstdint>

typedef __attribute__((ext_vector_type(16))) __bf16 v16bf;
typedef __attribute__((ext_vector_type(8)))  __bf16 v8bf;
typedef __attribute__((ext_vector_type(4)))  __bf16 v4bf;
typedef __attribute__((ext_vector_type(8)))  float  v8f;

#define LRELU_ALPHA 0.2f

#define CAT16(lo, hi) __builtin_shufflevector((lo), (hi), 0,1,2,3,4,5,6,7,8,9,10,11,12,13,14,15)

// ---------------------------------------------------------------------------
// 1) gather unique node embeddings, cast f32 -> bf16 (float4 in, 4x bf16 out)
//    Rows [U, Upad) are zero-filled so the padded GEMM tiles are benign.
// ---------------------------------------------------------------------------
__global__ void gather_cast_kernel(const float* __restrict__ nf,
                                   const int* __restrict__ uniq,
                                   __bf16* __restrict__ embB, int U, int Upad) {
  int idx = blockIdx.x * blockDim.x + threadIdx.x;   // over Upad*16 quads
  if (idx >= Upad * 16) return;
  int u = idx >> 4, q = idx & 15;
  v4bf o;
  if (u < U) {
    const float4 v = *(const float4*)(nf + (size_t)uniq[u] * 64 + q * 4);
    o[0] = (__bf16)v.x; o[1] = (__bf16)v.y; o[2] = (__bf16)v.z; o[3] = (__bf16)v.w;
  } else {
    o[0] = (__bf16)0.f; o[1] = (__bf16)0.f; o[2] = (__bf16)0.f; o[3] = (__bf16)0.f;
  }
  ((v4bf*)embB)[idx] = o;
}

// ---------------------------------------------------------------------------
// 2) pack W [H,64,64] f32 into bf16 B-fragment order:
//    WbT[(((h*4 + nt)*2 + s)*32 + lane)*16 + e] = W[h][s*32 + (e>>3)*16 + (lane>>4)*8 + (e&7)][nt*16 + (lane&15)]
//    so each lane's 16-element B fragment is 32 contiguous bytes.
// ---------------------------------------------------------------------------
__global__ void pack_w_kernel(const float* __restrict__ W,
                              __bf16* __restrict__ WbT, int n) {
  int o = blockIdx.x * blockDim.x + threadIdx.x;
  if (o >= n) return;
  int e    = o & 15;
  int lane = (o >> 4) & 31;
  int s    = (o >> 9) & 1;
  int nt   = (o >> 10) & 3;
  int h    = o >> 12;
  int K    = s * 32 + (e >> 3) * 16 + (lane >> 4) * 8 + (e & 7);
  int col  = nt * 16 + (lane & 15);
  WbT[o] = (__bf16)W[(size_t)h * 4096 + (size_t)K * 64 + col];
}

// ---------------------------------------------------------------------------
// 3) Wh[h,u,:] = embB[u,:] @ W[h]  via v_wmma_f32_16x16x32_bf16
//    One wave32 per (head, 16-row tile); U padded to Upad -> fully branchless:
//    A fragment = 2 contiguous b128 loads per k-step (ISA 7.12.2 16-bit A
//    layout: lane half 'hi' owns K runs [8hi,8hi+8) and [16+8hi,24+8hi)).
//    B fragment = 32 contiguous bytes from pre-packed WbT.
//    C: vgpr r -> M = r + 8*hi, N = nt*16 + (lane&15).
// ---------------------------------------------------------------------------
__global__ void wh_wmma_kernel(const __bf16* __restrict__ embB,
                               const __bf16* __restrict__ WbT,
                               float* __restrict__ Wh, int Upad, int H) {
  int wave  = (int)((blockIdx.x * blockDim.x + threadIdx.x) >> 5);
  int lane  = threadIdx.x & 31;
  int tiles = Upad >> 4;
  if (wave >= H * tiles) return;
  int h  = wave / tiles;
  int t  = wave % tiles;
  int m  = lane & 15;
  int hi = lane >> 4;
  int gm = t * 16 + m;

  // A fragments: row gm, chunks of 8 bf16 (16B each)
  const v8bf* ap = (const v8bf*)(embB + (size_t)gm * 64);
  v16bf a0 = CAT16(ap[hi],     ap[2 + hi]);   // K in {8hi..8hi+7, 16+8hi..23+8hi}
  v16bf a1 = CAT16(ap[4 + hi], ap[6 + hi]);   // +32

  float* out = Wh + ((size_t)h * Upad + (size_t)t * 16) * 64;
#pragma unroll
  for (int nt = 0; nt < 4; ++nt) {
    const v8bf* bp0 = (const v8bf*)(WbT + ((((size_t)h * 4 + nt) * 2 + 0) * 32 + lane) * 16);
    const v8bf* bp1 = (const v8bf*)(WbT + ((((size_t)h * 4 + nt) * 2 + 1) * 32 + lane) * 16);
    v16bf b0 = CAT16(bp0[0], bp0[1]);
    v16bf b1 = CAT16(bp1[0], bp1[1]);

    v8f c = {};
    c = __builtin_amdgcn_wmma_f32_16x16x32_bf16(false, a0, false, b0, (short)0, c, false, false);
    c = __builtin_amdgcn_wmma_f32_16x16x32_bf16(false, a1, false, b1, (short)0, c, false, false);

#pragma unroll
    for (int r = 0; r < 8; ++r)
      out[(size_t)(r + hi * 8) * 64 + nt * 16 + m] = c[r];
  }
}

// ---------------------------------------------------------------------------
// 4) per-(head,node) GAT score halves: wave per row, coalesced 64-wide load,
//    4 shuffle-reduced dot products against a_pos/a_neg halves.
// ---------------------------------------------------------------------------
__global__ void scores_kernel(const float* __restrict__ Wh,
                              const float* __restrict__ a_pos,
                              const float* __restrict__ a_neg,
                              float* __restrict__ spd, float* __restrict__ sps,
                              float* __restrict__ snd, float* __restrict__ sns,
                              int U, int Upad, int H) {
  int wave = (int)((blockIdx.x * blockDim.x + threadIdx.x) >> 5);
  int lane = threadIdx.x & 31;
  if (wave >= H * U) return;
  int h = wave / U;
  int u = wave - h * U;
  const float* w  = Wh + ((size_t)h * Upad + u) * 64;
  const float* ap = a_pos + (size_t)h * 128;
  const float* an = a_neg + (size_t)h * 128;
  float v0 = w[lane], v1 = w[lane + 32];
  float s0 = v0 * ap[lane]      + v1 * ap[lane + 32];
  float s1 = v0 * ap[lane + 64] + v1 * ap[lane + 96];
  float s2 = v0 * an[lane]      + v1 * an[lane + 32];
  float s3 = v0 * an[lane + 64] + v1 * an[lane + 96];
#pragma unroll
  for (int off = 16; off > 0; off >>= 1) {
    s0 += __shfl_xor(s0, off, 32);
    s1 += __shfl_xor(s1, off, 32);
    s2 += __shfl_xor(s2, off, 32);
    s3 += __shfl_xor(s3, off, 32);
  }
  if (lane == 0) { spd[wave] = s0; sps[wave] = s1; snd[wave] = s2; sns[wave] = s3; }
}

// ---------------------------------------------------------------------------
// 5) per-(row,head) wave: 2*DEG+1 edges -> leaky-relu -> softmax (shfl
//    reductions) -> weighted aggregation of Wh rows -> ReLU.  Exploits the
//    fixed edge layout: pos edges i*DEG.., self at N*DEG+i, neg edges i*DEG..
// ---------------------------------------------------------------------------
__global__ void aggregate_kernel(const float* __restrict__ Wh,
                                 const float* __restrict__ spd, const float* __restrict__ sps,
                                 const float* __restrict__ snd, const float* __restrict__ sns,
                                 const int* __restrict__ row_unique,
                                 const int* __restrict__ pos_col,
                                 const int* __restrict__ neg_col,
                                 float* __restrict__ out,
                                 int U, int Upad, int N, int H, int DEG) {
  int i    = blockIdx.x;
  int h    = threadIdx.x >> 5;
  int lane = threadIdx.x & 31;
  int E    = 2 * DEG + 1;            // pos + self + neg
  int hU   = h * U;

  int ru   = row_unique[i];
  float dp = spd[hU + ru];
  float dn = snd[hU + ru];

  auto edge = [&](int e, int& col, float& sc) {
    float s;
    if (e <= DEG) {                  // pos edges (incl. appended self edge)
      col = (e < DEG) ? pos_col[(size_t)i * DEG + e]
                      : pos_col[(size_t)N * DEG + i];
      s = dp + sps[hU + col];
    } else {                         // neg edges
      col = neg_col[(size_t)i * DEG + (e - DEG - 1)];
      s = dn + sns[hU + col];
    }
    sc = (s > 0.f) ? s : LRELU_ALPHA * s;
  };

  const float NEG_INF = -3.0e38f;
  int col0 = 0, col1 = 0;
  float sc0 = NEG_INF, sc1 = NEG_INF;
  if (lane < E)      edge(lane,      col0, sc0);
  if (lane + 32 < E) edge(lane + 32, col1, sc1);

  float mx = fmaxf(sc0, sc1);
#pragma unroll
  for (int off = 16; off > 0; off >>= 1) mx = fmaxf(mx, __shfl_xor(mx, off, 32));

  float ex0 = (lane < E)      ? __expf(sc0 - mx) : 0.f;
  float ex1 = (lane + 32 < E) ? __expf(sc1 - mx) : 0.f;
  float sm = ex0 + ex1;
#pragma unroll
  for (int off = 16; off > 0; off >>= 1) sm += __shfl_xor(sm, off, 32);
  float inv  = 1.f / sm;
  float att0 = ex0 * inv, att1 = ex1 * inv;

  const float* whh = Wh + (size_t)h * Upad * 64;
  float acc0 = 0.f, acc1 = 0.f;
  for (int e = 0; e < E; ++e) {
    float a = (e < 32) ? __shfl(att0, e, 32) : __shfl(att1, e - 32, 32);
    int   c = (e < 32) ? __shfl(col0, e, 32) : __shfl(col1, e - 32, 32);
    const float* src = whh + (size_t)c * 64;
    acc0 += a * src[lane];
    acc1 += a * src[lane + 32];
  }

  size_t base = (size_t)i * ((size_t)H * 64) + (size_t)h * 64;
  out[base + lane]      = fmaxf(acc0, 0.f);   // ReLU on concat
  out[base + lane + 32] = fmaxf(acc1, 0.f);
}

// ---------------------------------------------------------------------------
// 6) edge_out = edge_embedding [ET,64] @ edge_transform [64,64]  (tiny)
// ---------------------------------------------------------------------------
__global__ void edge_proj_kernel(const float* __restrict__ ee,
                                 const float* __restrict__ et,
                                 float* __restrict__ out, int ET) {
  int idx = blockIdx.x * blockDim.x + threadIdx.x;
  if (idx >= ET * 64) return;
  int r = idx >> 6, d = idx & 63;
  float acc = 0.f;
#pragma unroll 8
  for (int k = 0; k < 64; ++k) acc += ee[r * 64 + k] * et[k * 64 + d];
  out[idx] = acc;
}

// ---------------------------------------------------------------------------
extern "C" void kernel_launch(void* const* d_in, const int* in_sizes, int n_in,
                              void* d_out, int out_size, void* d_ws, size_t ws_size,
                              hipStream_t stream) {
  const float* node_features = (const float*)d_in[0];
  const float* W             = (const float*)d_in[1];
  const float* a_pos         = (const float*)d_in[2];
  const float* a_neg         = (const float*)d_in[3];
  const float* edge_emb      = (const float*)d_in[4];
  const float* edge_tr       = (const float*)d_in[5];
  const int*   uniq          = (const int*)d_in[6];
  const int*   row_unique    = (const int*)d_in[7];
  const int*   pos_col       = (const int*)d_in[9];
  const int*   neg_col       = (const int*)d_in[11];

  const int U    = in_sizes[6];                 // 50000
  const int Upad = (U + 15) & ~15;              // padded row count
  const int N    = in_sizes[7];                 // 20000
  const int H    = in_sizes[1] / (64 * 64);     // 4
  const int ET   = in_sizes[4] / 64;            // 32
  const int DEG  = in_sizes[10] / N;            // 16

  // workspace carve-out (256B aligned regions)
  char* wsp = (char*)d_ws;
  auto carve = [&](size_t bytes) {
    char* p = wsp;
    wsp += (bytes + 255) & ~(size_t)255;
    return p;
  };
  __bf16* embB = (__bf16*)carve((size_t)Upad * 64 * sizeof(__bf16));
  __bf16* WbT  = (__bf16*)carve((size_t)H * 64 * 64 * sizeof(__bf16));
  float*  Wh   = (float*)carve((size_t)H * Upad * 64 * sizeof(float));
  float*  spd  = (float*)carve((size_t)H * U * sizeof(float));
  float*  sps  = (float*)carve((size_t)H * U * sizeof(float));
  float*  snd  = (float*)carve((size_t)H * U * sizeof(float));
  float*  sns  = (float*)carve((size_t)H * U * sizeof(float));

  float* h_hidden = (float*)d_out;
  float* edge_out = (float*)d_out + (size_t)N * H * 64;

  { // 1) gather + cast (quad-vectorized, zero-fills pad rows)
    int total = Upad * 16;
    gather_cast_kernel<<<(total + 255) / 256, 256, 0, stream>>>(node_features, uniq, embB, U, Upad);
  }
  { // 2) pack W into per-lane B-fragment order
    int n = H * 64 * 64;
    pack_w_kernel<<<(n + 255) / 256, 256, 0, stream>>>(W, WbT, n);
  }
  { // 3) WMMA GEMM: one wave per (head, 16-row tile); 8 waves / block
    int tiles  = Upad / 16;
    int waves  = H * tiles;
    int blocks = (waves + 7) / 8;
    wh_wmma_kernel<<<blocks, 256, 0, stream>>>(embB, WbT, Wh, Upad, H);
  }
  { // 4) score scalars: wave per (h,u)
    int waves  = H * U;
    int blocks = (waves + 7) / 8;
    scores_kernel<<<blocks, 256, 0, stream>>>(Wh, a_pos, a_neg,
                                              spd, sps, snd, sns, U, Upad, H);
  }
  { // 5) softmax + aggregation: block per row, wave per head
    aggregate_kernel<<<N, H * 32, 0, stream>>>(Wh, spd, sps, snd, sns,
                                               row_unique, pos_col, neg_col,
                                               h_hidden, U, Upad, N, H, DEG);
  }
  { // 6) edge projection
    int total = ET * 64;
    edge_proj_kernel<<<(total + 255) / 256, 256, 0, stream>>>(edge_emb, edge_tr, edge_out, ET);
  }
}
